// NSAAttention_87419764342794
// MI455X (gfx1250) — matrix operand order repaired and checked by reading, not compiled, not run on hardware
//
// NSA attention for MI455X (gfx1250, wave32, WMMA bf16 16x16x32, f32 accum).
#include <hip/hip_runtime.h>
#include <stdint.h>

#define SEQ 1536
#define HQ 16
#define HKV 2
#define DH 128
#define GQA 8           // HQ / HKV
#define WIN 512
#define TC 95           // compressed tokens
#define TCP 96          // padded
#define NKB 24          // number of 64-token selection blocks (== pool outputs)
#define QK_SCALE 0.08838834764831845f  // 1/sqrt(128)

typedef __attribute__((ext_vector_type(16))) __bf16 v16bf;
typedef __attribute__((ext_vector_type(8)))  float  v8f;
typedef __attribute__((ext_vector_type(4)))  unsigned int u32x4;

union V16U { u32x4 q[2]; v16bf v; };

__device__ inline unsigned short f2bf(float f) {   // RNE f32 -> bf16
  unsigned u = __float_as_uint(f);
  u += 0x7FFFu + ((u >> 16) & 1u);
  return (unsigned short)(u >> 16);
}

// B operand: 16 contiguous bf16 (column of B for this lane), 16B-aligned chunks.
__device__ inline v16bf load_b16(const unsigned short* p) {
  V16U u;
  u.q[0] = *(const u32x4*)p;
  u.q[1] = *(const u32x4*)(p + 8);
  return u.v;
}
// A operand: two contiguous 8-bf16 runs (K=b..b+7 and K=b+8..b+15, b=16*(lane>=16)).
__device__ inline v16bf load_a2(const unsigned short* p0, const unsigned short* p1) {
  V16U u;
  u.q[0] = *(const u32x4*)p0;
  u.q[1] = *(const u32x4*)p1;
  return u.v;
}
__device__ inline v8f wmma_bf16(v16bf a, v16bf b, v8f c) {
  return __builtin_amdgcn_wmma_f32_16x16x32_bf16(false, a, false, b, (short)0, c,
                                                 false, false);
}

// ---------------- conversion: q,k -> bf16; v -> bf16 transposed [HKV][D][S] ----
__global__ void convert_kernel(const float* __restrict__ q, const float* __restrict__ k,
                               const float* __restrict__ v,
                               unsigned short* __restrict__ qbf,
                               unsigned short* __restrict__ kbf,
                               unsigned short* __restrict__ vtr) {
  int i = blockIdx.x * blockDim.x + threadIdx.x;
  if (i < SEQ * HQ * DH) qbf[i] = f2bf(q[i]);
  if (i < SEQ * HKV * DH) {
    kbf[i] = f2bf(k[i]);
    int s = i / (HKV * DH);
    int rem = i - s * (HKV * DH);
    int h = rem / DH, d = rem - h * DH;
    vtr[(h * DH + d) * SEQ + s] = f2bf(v[i]);
  }
}

__global__ void zero_kernel(float* p, int n) {
  int i = blockIdx.x * blockDim.x + threadIdx.x;
  if (i < n) p[i] = 0.f;
}

// ---------------- compression: blk(32x128) @ W(4096x128) + b ------------------
__global__ void compress_kernel(const float* __restrict__ x,   // [S][HKV][D]
                                const float* __restrict__ W,   // [32*D][D]
                                const float* __restrict__ b,   // [D]
                                const float* __restrict__ pe,  // [32][D]
                                unsigned short* __restrict__ out,
                                int transposed) {
  const int tc = blockIdx.x;   // 0..TCP-1 (95 = zero pad)
  const int h  = blockIdx.y;
  const int t  = threadIdx.x;  // 0..127 = output d
  __shared__ float xs[32 * DH];
  float acc = 0.f;
  if (tc < TC) {
    for (int i = t; i < 32 * DH; i += 128) {
      int j = i >> 7, d = i & 127;
      xs[i] = x[((tc * 16 + j) * HKV + h) * DH + d] + pe[i];
    }
    __syncthreads();
    acc = b[t];
    for (int i = 0; i < 32 * DH; ++i) acc += xs[i] * W[i * DH + t];
  }
  unsigned short r = f2bf(acc);
  if (transposed) out[(h * DH + t) * TCP + tc] = r;   // cvtr [HKV][D][TCP]
  else            out[(h * TCP + tc) * DH + t] = r;   // ckbf [HKV][TCP][D]
}

// ---------------- gates: sigmoid(q @ Wg + bg) ---------------------------------
__global__ void gate_kernel(const float* __restrict__ q, const float* __restrict__ Wg,
                            const float* __restrict__ bg, float* __restrict__ gate) {
  int i = blockIdx.x * blockDim.x + threadIdx.x;  // s*HQ + h
  if (i >= SEQ * HQ) return;
  const float* qp = q + i * DH;
  float g0 = bg[0], g1 = bg[1], g2 = bg[2];
  for (int d = 0; d < DH; ++d) {
    float qv = qp[d];
    g0 += qv * Wg[d * 3 + 0];
    g1 += qv * Wg[d * 3 + 1];
    g2 += qv * Wg[d * 3 + 2];
  }
  gate[i * 3 + 0] = 1.f / (1.f + __expf(-g0));
  gate[i * 3 + 1] = 1.f / (1.f + __expf(-g1));
  gate[i * 3 + 2] = 1.f / (1.f + __expf(-g2));
}

// ---------------- stage 1: compressed-KV attention + score accumulation ------
__global__ __launch_bounds__(32) void stage1_kernel(
    const unsigned short* __restrict__ qbf,   // [S][HQ][D] bf16
    const unsigned short* __restrict__ ckbf,  // [HKV][TCP][D] bf16
    const unsigned short* __restrict__ cvtr,  // [HKV][D][TCP] bf16
    float* __restrict__ score,                // [HKV][S][TCP] (zeroed)
    float* __restrict__ cmp_o) {              // [S][HQ][D]
  const int qt = blockIdx.x, h = blockIdx.y;
  const int hkv = h / GQA;
  const int s0 = qt * 16;
  const int lane = threadIdx.x, nn = lane & 15, hl = lane >> 4;
  __shared__ __align__(16) unsigned short pds[16 * 32];

  v8f c[6];
#pragma unroll
  for (int tt = 0; tt < 6; ++tt) c[tt] = v8f{};
#pragma unroll
  for (int dc = 0; dc < 4; ++dc) {
    const unsigned short* qb = qbf + ((s0 + nn) * HQ + h) * DH + dc * 32 + 16 * hl;
    v16bf a = load_a2(qb, qb + 8);
#pragma unroll
    for (int tt = 0; tt < 6; ++tt) {
      const unsigned short* kb =
          ckbf + (hkv * TCP + tt * 16 + nn) * DH + dc * 32 + 16 * hl;
      c[tt] = wmma_bf16(a, load_b16(kb), c[tt]);
    }
  }
  float p[6][8];
  bool  mk[6][8];
  float l[8];
#pragma unroll
  for (int r = 0; r < 8; ++r) {
    const int srow = s0 + r + 8 * hl;
    float vmx = -1e30f;
#pragma unroll
    for (int tt = 0; tt < 6; ++tt) {
      int tcol = tt * 16 + nn;
      bool ok = (tcol < TC) && (tcol * 16 + 31 <= srow);
      mk[tt][r] = ok;
      p[tt][r] = ok ? c[tt][r] * QK_SCALE : -1e30f;
      vmx = fmaxf(vmx, p[tt][r]);
    }
#pragma unroll
    for (int i = 1; i < 16; i <<= 1) vmx = fmaxf(vmx, __shfl_xor(vmx, i, 32));
    float rs = 0.f;
#pragma unroll
    for (int tt = 0; tt < 6; ++tt) {
      p[tt][r] = mk[tt][r] ? __expf(p[tt][r] - vmx) : 0.f;
      rs += p[tt][r];
    }
#pragma unroll
    for (int i = 1; i < 16; i <<= 1) rs += __shfl_xor(rs, i, 32);
    l[r] = rs;
  }
  float dinv[8];
#pragma unroll
  for (int r = 0; r < 8; ++r) dinv[r] = 1.f / fmaxf(l[r], 1e-9f);

  // normalized probs summed over the GQA group -> pooled-score input
#pragma unroll
  for (int tt = 0; tt < 6; ++tt)
#pragma unroll
    for (int r = 0; r < 8; ++r)
      if (mk[tt][r]) {
        const int srow = s0 + r + 8 * hl;
        atomicAdd(&score[(hkv * SEQ + srow) * TCP + tt * 16 + nn], p[tt][r] * dinv[r]);
      }

  v8f acc[8];
#pragma unroll
  for (int nt = 0; nt < 8; ++nt) acc[nt] = v8f{};
  for (int pr = 0; pr < 3; ++pr) {
    __syncthreads();
#pragma unroll
    for (int r = 0; r < 8; ++r) {
      int M = r + 8 * hl;
      pds[M * 32 + nn]      = f2bf(p[2 * pr][r]);
      pds[M * 32 + 16 + nn] = f2bf(p[2 * pr + 1][r]);
    }
    __syncthreads();
    v16bf pa = load_a2(&pds[nn * 32 + 16 * hl], &pds[nn * 32 + 16 * hl + 8]);
#pragma unroll
    for (int nt = 0; nt < 8; ++nt) {
      const unsigned short* vb =
          cvtr + (hkv * DH + nt * 16 + nn) * TCP + pr * 32 + 16 * hl;
      acc[nt] = wmma_bf16(pa, load_b16(vb), acc[nt]);
    }
  }
#pragma unroll
  for (int r = 0; r < 8; ++r) {
    const int srow = s0 + r + 8 * hl;
#pragma unroll
    for (int nt = 0; nt < 8; ++nt)
      cmp_o[(srow * HQ + h) * DH + nt * 16 + nn] = acc[nt][r] * dinv[r];
  }
}

// ---------------- stage 2: avg-pool + stable top-16 -> 24-bit block mask ------
__global__ void topk_kernel(const float* __restrict__ score,
                            unsigned int* __restrict__ selmask) {
  int i = blockIdx.x * blockDim.x + threadIdx.x;  // hkv*SEQ + s
  if (i >= HKV * SEQ) return;
  const float* sc = score + i * TCP;
  float pooled[NKB];
  for (int o = 0; o < NKB; ++o) {
    float sum = 0.f;
    int cnt = 0;
    for (int j = 0; j < 5; ++j) {
      int w = 4 * o + j;
      if (w < TC) { sum += sc[w]; cnt++; }
    }
    pooled[o] = sum / (float)cnt;
  }
  unsigned taken = 0;
  for (int it = 0; it < 16; ++it) {
    float best = 0.f;
    int bi = 0;
    bool found = false;
    for (int j = 0; j < NKB; ++j) {
      if ((taken >> j) & 1u) continue;
      if (!found || pooled[j] > best) { best = pooled[j]; bi = j; found = true; }
    }
    taken |= (1u << bi);
  }
  selmask[i] = taken;
}

// ---------------- stage 3: block-sparse + window attention, gated combine -----
__global__ __launch_bounds__(32) void stage3_kernel(
    const unsigned short* __restrict__ qbf,   // [S][HQ][D]
    const unsigned short* __restrict__ kbf,   // [S][HKV][D]
    const unsigned short* __restrict__ vtr,   // [HKV][D][S]
    const unsigned int*  __restrict__ selmask,// [HKV][S]
    const float* __restrict__ gate,           // [S][HQ][3]
    const float* __restrict__ cmp_o,          // [S][HQ][D]
    float* __restrict__ out) {                // [S][HQ][D]
  const int qt = blockIdx.x, h = blockIdx.y;
  const int hkv = h / GQA;
  const int s0 = qt * 16;
  const int lane = threadIdx.x, nn = lane & 15, hl = lane >> 4;
  __shared__ __align__(16) unsigned short pds[16 * 32];

  v16bf qa[4];
#pragma unroll
  for (int dc = 0; dc < 4; ++dc) {
    const unsigned short* base = qbf + ((s0 + nn) * HQ + h) * DH + dc * 32 + 16 * hl;
    qa[dc] = load_a2(base, base + 8);
  }
  unsigned selm[8], selun = 0;
#pragma unroll
  for (int r = 0; r < 8; ++r) {
    selm[r] = selmask[hkv * SEQ + s0 + r + 8 * hl];
    selun |= selm[r];
  }
  selun |= __shfl_xor(selun, 16, 32);

  float msel[8], lsel[8], mwin[8], lwin[8];
  v8f accsel[8], accwin[8];
#pragma unroll
  for (int r = 0; r < 8; ++r) { msel[r] = -1e30f; lsel[r] = 0.f; mwin[r] = -1e30f; lwin[r] = 0.f; }
#pragma unroll
  for (int nt = 0; nt < 8; ++nt) { accsel[nt] = v8f{}; accwin[nt] = v8f{}; }

  const int nsteps = (s0 + 16 + 31) / 32;
  for (int st = 0; st < nsteps; ++st) {
    const int u0 = st * 32;
    const int blk = u0 >> 6;
    const bool anysel = (selun >> blk) & 1u;
    const bool anywin = (u0 + 31) > (s0 - WIN);
    if (!anysel && !anywin) continue;

    v8f c0 = v8f{}, c1 = v8f{};
#pragma unroll
    for (int dc = 0; dc < 4; ++dc) {
      const unsigned short* kb0 = kbf + ((u0 + nn) * HKV + hkv) * DH + dc * 32 + 16 * hl;
      const unsigned short* kb1 = kbf + ((u0 + 16 + nn) * HKV + hkv) * DH + dc * 32 + 16 * hl;
      c0 = wmma_bf16(qa[dc], load_b16(kb0), c0);
      c1 = wmma_bf16(qa[dc], load_b16(kb1), c1);
    }
    const int u0n = u0 + nn, u1n = u0 + 16 + nn;
    float lg0[8], lg1[8];
#pragma unroll
    for (int r = 0; r < 8; ++r) { lg0[r] = c0[r] * QK_SCALE; lg1[r] = c1[r] * QK_SCALE; }

    auto process = [&](float* m, float* l, v8f* acc, bool isSel) {
      float p0[8], p1[8], fac[8];
#pragma unroll
      for (int r = 0; r < 8; ++r) {
        const int srow = s0 + r + 8 * hl;
        bool m0 = u0n <= srow, m1 = u1n <= srow;
        if (isSel) {
          bool sb = (selm[r] >> blk) & 1u;
          m0 = m0 && sb;
          m1 = m1 && sb;
        } else {
          m0 = m0 && (u0n > srow - WIN);
          m1 = m1 && (u1n > srow - WIN);
        }
        float a0 = m0 ? lg0[r] : -1e30f;
        float a1 = m1 ? lg1[r] : -1e30f;
        float vmx = fmaxf(a0, a1);
#pragma unroll
        for (int i = 1; i < 16; i <<= 1) vmx = fmaxf(vmx, __shfl_xor(vmx, i, 32));
        float mn = fmaxf(m[r], vmx);
        fac[r] = __expf(m[r] - mn);
        m[r] = mn;
        p0[r] = m0 ? __expf(lg0[r] - mn) : 0.f;
        p1[r] = m1 ? __expf(lg1[r] - mn) : 0.f;
        float rs = p0[r] + p1[r];
#pragma unroll
        for (int i = 1; i < 16; i <<= 1) rs += __shfl_xor(rs, i, 32);
        l[r] = l[r] * fac[r] + rs;
      }
#pragma unroll
      for (int nt = 0; nt < 8; ++nt)
#pragma unroll
        for (int r = 0; r < 8; ++r) acc[nt][r] *= fac[r];
      __syncthreads();
#pragma unroll
      for (int r = 0; r < 8; ++r) {
        int M = r + 8 * hl;
        pds[M * 32 + nn]      = f2bf(p0[r]);
        pds[M * 32 + 16 + nn] = f2bf(p1[r]);
      }
      __syncthreads();
      v16bf pa = load_a2(&pds[nn * 32 + 16 * hl], &pds[nn * 32 + 16 * hl + 8]);
#pragma unroll
      for (int nt = 0; nt < 8; ++nt) {
        const unsigned short* vb = vtr + (hkv * DH + nt * 16 + nn) * SEQ + u0 + 16 * hl;
        acc[nt] = wmma_bf16(pa, load_b16(vb), acc[nt]);
      }
    };
    if (anysel) process(msel, lsel, accsel, true);
    if (anywin) process(mwin, lwin, accwin, false);
  }

#pragma unroll
  for (int r = 0; r < 8; ++r) {
    const int srow = s0 + r + 8 * hl;
    const float* gp = gate + (srow * HQ + h) * 3;
    float g0 = gp[0], g1 = gp[1], g2 = gp[2];
    float ds = 1.f / fmaxf(lsel[r], 1e-9f);
    float dw = 1.f / fmaxf(lwin[r], 1e-9f);
#pragma unroll
    for (int nt = 0; nt < 8; ++nt) {
      int d = nt * 16 + nn;
      float cm = cmp_o[(srow * HQ + h) * DH + d];
      out[(srow * HQ + h) * DH + d] =
          g0 * (accsel[nt][r] * ds) + g1 * (accwin[nt][r] * dw) + g2 * cm;
    }
  }
}

extern "C" void kernel_launch(void* const* d_in, const int* in_sizes, int n_in,
                              void* d_out, int out_size, void* d_ws, size_t ws_size,
                              hipStream_t stream) {
  (void)in_sizes; (void)n_in; (void)out_size; (void)ws_size;
  const float* q   = (const float*)d_in[0];
  const float* k   = (const float*)d_in[1];
  const float* v   = (const float*)d_in[2];
  const float* Wk  = (const float*)d_in[3];
  const float* bk  = (const float*)d_in[4];
  const float* Wv  = (const float*)d_in[5];
  const float* bv  = (const float*)d_in[6];
  const float* pek = (const float*)d_in[7];
  const float* pev = (const float*)d_in[8];
  const float* Wg  = (const float*)d_in[9];
  const float* bg  = (const float*)d_in[10];
  float* out = (float*)d_out;

  char* ws = (char*)d_ws;
  size_t off = 0;
  auto alloc = [&](size_t bytes) {
    void* p = ws + off;
    off = (off + bytes + 255) & ~(size_t)255;
    return p;
  };
  unsigned short* qbf  = (unsigned short*)alloc((size_t)SEQ * HQ * DH * 2);
  unsigned short* kbf  = (unsigned short*)alloc((size_t)SEQ * HKV * DH * 2);
  unsigned short* vtr  = (unsigned short*)alloc((size_t)SEQ * HKV * DH * 2);
  unsigned short* ckbf = (unsigned short*)alloc((size_t)HKV * TCP * DH * 2);
  unsigned short* cvtr = (unsigned short*)alloc((size_t)HKV * TCP * DH * 2);
  float* score        = (float*)alloc((size_t)HKV * SEQ * TCP * 4);
  unsigned int* selmk = (unsigned int*)alloc((size_t)HKV * SEQ * 4);
  float* gatep        = (float*)alloc((size_t)SEQ * HQ * 3 * 4);
  float* cmp_o        = (float*)alloc((size_t)SEQ * HQ * DH * 4);

  convert_kernel<<<(SEQ * HQ * DH + 255) / 256, 256, 0, stream>>>(q, k, v, qbf, kbf, vtr);
  zero_kernel<<<(HKV * SEQ * TCP + 255) / 256, 256, 0, stream>>>(score, HKV * SEQ * TCP);
  compress_kernel<<<dim3(TCP, HKV), 128, 0, stream>>>(k, Wk, bk, pek, ckbf, 0);
  compress_kernel<<<dim3(TCP, HKV), 128, 0, stream>>>(v, Wv, bv, pev, cvtr, 1);
  gate_kernel<<<(SEQ * HQ + 127) / 128, 128, 0, stream>>>(q, Wg, bg, gatep);
  stage1_kernel<<<dim3(SEQ / 16, HQ), 32, 0, stream>>>(qbf, ckbf, cvtr, score, cmp_o);
  topk_kernel<<<(HKV * SEQ + 127) / 128, 128, 0, stream>>>(score, selmk);
  stage3_kernel<<<dim3(SEQ / 16, HQ), 32, 0, stream>>>(qbf, kbf, vtr, selmk, gatep,
                                                       cmp_o, out);
}
